// Informer_19181323944433
// MI455X (gfx1250) — matrix operand order repaired and checked by reading, not compiled, not run on hardware
//
#include <hip/hip_runtime.h>
#include <math.h>

// ---------------------------------------------------------------------------
// Informer forward for MI455X (gfx1250, wave32, WMMA).
// All dense projections run through a bf16 WMMA GEMM templated on (K,N) so the
// K-loop fully unrolls (v_wmma_f32_16x16x32_bf16 chains) and each wave computes
// a 16x64 tile: one A fragment feeds 4 WMMAs. Workload is HBM-bound
// (AI ~30 FLOP/B vs 23.3 TB/s), so passes over activations are minimized:
// fused K|V projection, BN folded into conv weights, epilogue-fused
// bias/residual/GELU/ELU, float4 (b128) loads on aligned 32B segments.
// ---------------------------------------------------------------------------

#define BATCH  32
#define DMODEL 64
#define NHEADS 8
#define DFF    128
#define CIN    8
#define LENC   4096
#define LDEC   2048
#define MAXTOP 9

typedef __attribute__((ext_vector_type(16))) __bf16 v16bf;
typedef __attribute__((ext_vector_type(8)))  float  v8f;

// ---------------- WMMA GEMM:  Y = act(X @ W^T + bias [+ resid]) -------------
// X: T x K fp32 row-major, W: N x K fp32 row-major (torch-style), Y: T x N.
// act: 0=none, 1=exact GELU, 2=ELU. One wave -> one 16x64 output tile
// (4 accumulators reuse a single A fragment per 32-wide K chunk).
template <int K, int N>
__global__ void __launch_bounds__(256) gemm_wmma_k(
    const float* __restrict__ X, const float* __restrict__ W,
    const float* __restrict__ bias, const float* __restrict__ resid,
    float* __restrict__ Y, int T, int act)
{
  constexpr int NG = N / 64;             // 64-column groups
  int gtid  = blockIdx.x * blockDim.x + threadIdx.x;
  int wave  = gtid >> 5;
  int lane  = threadIdx.x & 31;
  int mtile = wave / NG;
  int ng    = wave - mtile * NG;
  if (mtile * 16 >= T) return;          // wave-uniform: EXEC all-1 for WMMA
  int row0 = mtile << 4;
  int col0 = ng * 64;
  int m  = lane & 15;
  int kh = lane >> 4;

  v8f acc[4] = {};
  const float* xrow = X + (size_t)(row0 + m) * K;
#pragma unroll
  for (int kc = 0; kc < K; kc += 32) {
    // A fragment (16x32 bf16): lane m (m<16) holds row m, K 0..15;
    // lane m+16 holds row m, K 16..31.  32B-aligned float4 loads.
    const float4* a0 = (const float4*)(xrow + kc + (kh << 4));
    const float4* a1 = (const float4*)(xrow + kc + 8 + (kh << 4));
    float4 s0 = a0[0], s1 = a0[1], s2 = a1[0], s3 = a1[1];
    v16bf a;
    a[0]  = (__bf16)s0.x; a[1]  = (__bf16)s0.y; a[2]  = (__bf16)s0.z; a[3]  = (__bf16)s0.w;
    a[4]  = (__bf16)s1.x; a[5]  = (__bf16)s1.y; a[6]  = (__bf16)s1.z; a[7]  = (__bf16)s1.w;
    a[8]  = (__bf16)s2.x; a[9]  = (__bf16)s2.y; a[10] = (__bf16)s2.z; a[11] = (__bf16)s2.w;
    a[12] = (__bf16)s3.x; a[13] = (__bf16)s3.y; a[14] = (__bf16)s3.z; a[15] = (__bf16)s3.w;
#pragma unroll
    for (int nt = 0; nt < 4; ++nt) {
      // B fragment (32x16 bf16): lane n holds column n, K 0..15 (lanes 0-15)
      // or K 16..31 (lanes 16-31). B[k][n] = W[n][k] -> contiguous W row.
      const float4* w4 = (const float4*)(W + (size_t)(col0 + nt * 16 + m) * K
                                           + kc + (kh << 4));
      float4 w0 = w4[0], w1 = w4[1], w2 = w4[2], w3 = w4[3];
      v16bf bm;
      bm[0]  = (__bf16)w0.x; bm[1]  = (__bf16)w0.y; bm[2]  = (__bf16)w0.z; bm[3]  = (__bf16)w0.w;
      bm[4]  = (__bf16)w1.x; bm[5]  = (__bf16)w1.y; bm[6]  = (__bf16)w1.z; bm[7]  = (__bf16)w1.w;
      bm[8]  = (__bf16)w2.x; bm[9]  = (__bf16)w2.y; bm[10] = (__bf16)w2.z; bm[11] = (__bf16)w2.w;
      bm[12] = (__bf16)w3.x; bm[13] = (__bf16)w3.y; bm[14] = (__bf16)w3.z; bm[15] = (__bf16)w3.w;
      acc[nt] = __builtin_amdgcn_wmma_f32_16x16x32_bf16(false, a, false, bm,
                                                        (short)0, acc[nt], false, false);
    }
  }
  // C/D layout: VGPR r of lane n -> (M = r + 8*(lane>=16), N = n)
#pragma unroll
  for (int nt = 0; nt < 4; ++nt) {
    int col = col0 + nt * 16 + m;
    float bv = bias ? bias[col] : 0.0f;
#pragma unroll
    for (int r = 0; r < 8; ++r) {
      int row = row0 + r + (kh << 3);
      float v = acc[nt][r] + bv;
      if (resid) v += resid[(size_t)row * N + col];
      if (act == 1)      v = 0.5f * v * (1.0f + erff(v * 0.70710678118f));
      else if (act == 2) v = (v > 0.0f) ? v : (__expf(v) - 1.0f);
      Y[(size_t)row * N + col] = v;
    }
  }
}

// ---------------- LayerNorm over last dim (64), one wave per token ----------
__global__ void __launch_bounds__(256) layernorm_k(
    const float* __restrict__ X, const float* __restrict__ g,
    const float* __restrict__ b, float* __restrict__ Y, int T)
{
  int wid  = (blockIdx.x * blockDim.x + threadIdx.x) >> 5;
  int lane = threadIdx.x & 31;
  if (wid >= T) return;
  const float* x = X + (size_t)wid * DMODEL;
  float a0 = x[lane], a1 = x[lane + 32];
  float s = a0 + a1;
  for (int o = 16; o > 0; o >>= 1) s += __shfl_xor(s, o, 32);
  float mean = s * (1.0f / 64.0f);
  float d0 = a0 - mean, d1 = a1 - mean;
  float vv = d0 * d0 + d1 * d1;
  for (int o = 16; o > 0; o >>= 1) vv += __shfl_xor(vv, o, 32);
  float inv = rsqrtf(vv * (1.0f / 64.0f) + 1e-5f);
  float* y = Y + (size_t)wid * DMODEL;
  y[lane]      = d0 * inv * g[lane]      + b[lane];
  y[lane + 32] = d1 * inv * g[lane + 32] + b[lane + 32];
}

// ---------------- Data embedding: circular conv1d(C_IN->64,k=3) + pos -------
__global__ void embed_k(const float* __restrict__ x, const float* __restrict__ w,
                        float* __restrict__ y, int L)
{
  int idx = blockIdx.x * blockDim.x + threadIdx.x;
  int total = BATCH * L * DMODEL;
  if (idx >= total) return;
  int o = idx & 63;
  int t = idx >> 6;
  int l = t % L;
  int b = t / L;
  const float* xb = x + (size_t)b * L * CIN;
  int lm = (l == 0) ? (L - 1) : (l - 1);
  int lp = (l == L - 1) ? 0 : (l + 1);
  float acc = 0.0f;
#pragma unroll
  for (int c = 0; c < CIN; ++c) {
    const float* wr = w + (o * CIN + c) * 3;
    acc += xb[(size_t)lm * CIN + c] * wr[0]
         + xb[(size_t)l  * CIN + c] * wr[1]
         + xb[(size_t)lp * CIN + c] * wr[2];
  }
  float freq = __expf(-(float)(o & ~1) * (9.210340372f / 64.0f)); // ln(1e4)/64
  float ang  = (float)l * freq;
  acc += (o & 1) ? __cosf(ang) : __sinf(ang);
  y[idx] = acc;
}

// ---------------- Pack [wk; wv] -> (128 x 64) fused KV weight ---------------
__global__ void pack_kv_k(const float* __restrict__ wk, const float* __restrict__ wv,
                          const float* __restrict__ bk, const float* __restrict__ bv,
                          float* __restrict__ wkv, float* __restrict__ bkv)
{
  int i = blockIdx.x * blockDim.x + threadIdx.x;
  if (i < 4096)        wkv[i] = wk[i];
  else if (i < 8192)   wkv[i] = wv[i - 4096];
  if (i < 64)          bkv[i] = bk[i];
  else if (i < 128)    bkv[i] = bv[i - 64];
}

// ---------------- Fold BN into conv weights: wc (64 x 192), bc (64) ---------
__global__ void pack_conv_k(const float* __restrict__ w, const float* __restrict__ bconv,
                            const float* __restrict__ g, const float* __restrict__ be,
                            const float* __restrict__ rm, const float* __restrict__ rv,
                            float* __restrict__ wc, float* __restrict__ bc)
{
  int i = blockIdx.x * blockDim.x + threadIdx.x;
  if (i >= 64 * 192) return;
  int o = i / 192; int k = i - o * 192;
  int s = k >> 6;  int c = k & 63;
  float scale = g[o] * rsqrtf(rv[o] + 1e-5f);
  wc[i] = w[((size_t)o * 64 + c) * 3 + s] * scale;
  if (i < 64) {
    float s0 = g[i] * rsqrtf(rv[i] + 1e-5f);
    bc[i] = bconv[i] * s0 + be[i] - rm[i] * s0;
  }
}

// ---------------- im2col (k=3, circular): (B,L,64) -> (B,L,192) -------------
__global__ void im2col3_k(const float* __restrict__ x, float* __restrict__ y, int L)
{
  int idx = blockIdx.x * blockDim.x + threadIdx.x;
  int total = BATCH * L * 192;
  if (idx >= total) return;
  int c = idx % 192;
  int t = idx / 192;
  int l = t % L;
  int b = t / L;
  int s = c / 64, ch = c & 63;
  int ls = l + s - 1;
  ls = (ls < 0) ? (L - 1) : ((ls >= L) ? 0 : ls);
  y[idx] = x[((size_t)b * L + ls) * 64 + ch];
}

// ---------------- maxpool k=3 s=2 pad=-inf over length ---------------------
__global__ void maxpool3s2_k(const float* __restrict__ y, float* __restrict__ o, int L)
{
  int Lo = L >> 1;
  int idx = blockIdx.x * blockDim.x + threadIdx.x;
  int total = BATCH * Lo * 64;
  if (idx >= total) return;
  int d = idx & 63;
  int t = idx >> 6;
  int j = t % Lo;
  int b = t / Lo;
  float m = -3.4e38f;
#pragma unroll
  for (int w = 0; w < 3; ++w) {
    int l = 2 * j - 1 + w;
    if (l >= 0 && l < L) m = fmaxf(m, y[((size_t)b * L + l) * 64 + d]);
  }
  o[idx] = m;
}

// ---------------- Prob-sparse sparsity measure M ----------------------------
__global__ void mscore_k(const float* __restrict__ Qb, const float* __restrict__ KVb,
                         const int* __restrict__ idx, float* __restrict__ M,
                         int LQ, int LK, int ns)
{
  int gid = blockIdx.x * blockDim.x + threadIdx.x;
  int total = BATCH * NHEADS * LQ;
  if (gid >= total) return;
  int l = gid % LQ;
  int h = (gid / LQ) % NHEADS;
  int b = gid / (LQ * NHEADS);
  const float* q = Qb + ((size_t)b * LQ + l) * 64 + h * 8;
  float qv[8];
#pragma unroll
  for (int e = 0; e < 8; ++e) qv[e] = q[e];
  float mx = -3.4e38f, sm = 0.0f;
  for (int s = 0; s < ns; ++s) {
    int ki = idx[l * ns + s];
    const float* k = KVb + ((size_t)b * LK + ki) * 128 + h * 8;
    float d = 0.0f;
#pragma unroll
    for (int e = 0; e < 8; ++e) d += qv[e] * k[e];
    mx = fmaxf(mx, d);
    sm += d;
  }
  M[gid] = mx - sm / (float)LK;
}

// ---------------- top-k (k<=9), one wave per (b,h) --------------------------
__global__ void topk_k(const float* __restrict__ M, int* __restrict__ mtop,
                       int LQ, int ntop)
{
  __shared__ float sval[32 * MAXTOP];
  __shared__ int   sidx[32 * MAXTOP];
  const float* m = M + (size_t)blockIdx.x * LQ;
  int lane = threadIdx.x;
  float val[MAXTOP]; int vid[MAXTOP];
#pragma unroll
  for (int i = 0; i < MAXTOP; ++i) { val[i] = -3.4e38f; vid[i] = 0x7fffffff; }
  for (int l = lane; l < LQ; l += 32) {
    float v = m[l];
    if (v > val[ntop - 1]) {
      int p = ntop - 1;
      while (p > 0 && val[p - 1] < v) { val[p] = val[p - 1]; vid[p] = vid[p - 1]; --p; }
      val[p] = v; vid[p] = l;
    }
  }
  for (int i = 0; i < ntop; ++i) { sval[lane * ntop + i] = val[i]; sidx[lane * ntop + i] = vid[i]; }
  __syncthreads();
  if (lane == 0) {
    int* out = mtop + blockIdx.x * ntop;
    unsigned char used[32 * MAXTOP];
    for (int j = 0; j < 32 * ntop; ++j) used[j] = 0;
    for (int r = 0; r < ntop; ++r) {
      float best = -3.4e38f; int bj = 0; int bidx = 0x7fffffff;
      for (int j = 0; j < 32 * ntop; ++j) {
        if (!used[j]) {
          float v = sval[j];
          if (v > best || (v == best && sidx[j] < bidx)) { best = v; bidx = sidx[j]; bj = j; }
        }
      }
      used[bj] = 1;
      out[r] = bidx;
    }
  }
}

// ---------------- attention for the ntop active queries (online softmax) ----
__global__ void __launch_bounds__(256) probattn_k(
    const float* __restrict__ Qb, const float* __restrict__ KVb,
    const int* __restrict__ mtop, float* __restrict__ upd,
    int LQ, int LK, int ntop, int maskflag)
{
  int wid  = (blockIdx.x * blockDim.x + threadIdx.x) >> 5;
  int lane = threadIdx.x & 31;
  int total = BATCH * NHEADS * ntop;
  if (wid >= total) return;
  int u = wid % ntop;
  int h = (wid / ntop) % NHEADS;
  int b = wid / (ntop * NHEADS);
  int qi = mtop[(b * NHEADS + h) * ntop + u];
  const float* q = Qb + ((size_t)b * LQ + qi) * 64 + h * 8;
  const float scale = 0.35355339059f;  // 1/sqrt(8)
  float qv[8];
#pragma unroll
  for (int e = 0; e < 8; ++e) qv[e] = q[e] * scale;
  float mi = -3.4e38f, li = 0.0f, acc[8];
#pragma unroll
  for (int e = 0; e < 8; ++e) acc[e] = 0.0f;
  int kmax = maskflag ? (qi + 1) : LK;
  for (int s = lane; s < kmax; s += 32) {
    const float* kp = KVb + ((size_t)b * LK + s) * 128 + h * 8;
    float d = 0.0f;
#pragma unroll
    for (int e = 0; e < 8; ++e) d += qv[e] * kp[e];
    float nm = fmaxf(mi, d);
    float sc = __expf(mi - nm);
    float w  = __expf(d - nm);
    li = li * sc + w;
    const float* vp = kp + 64;
#pragma unroll
    for (int e = 0; e < 8; ++e) acc[e] = acc[e] * sc + w * vp[e];
    mi = nm;
  }
  for (int o = 16; o > 0; o >>= 1) {
    float mo = __shfl_xor(mi, o, 32);
    float lo = __shfl_xor(li, o, 32);
    float ao[8];
#pragma unroll
    for (int e = 0; e < 8; ++e) ao[e] = __shfl_xor(acc[e], o, 32);
    float nm = fmaxf(mi, mo);
    float s1 = __expf(mi - nm), s2 = __expf(mo - nm);
    li = li * s1 + lo * s2;
#pragma unroll
    for (int e = 0; e < 8; ++e) acc[e] = acc[e] * s1 + ao[e] * s2;
    mi = nm;
  }
  if (lane == 0) {
    float invl = 1.0f / li;
    float* o = upd + (size_t)wid * 8;
#pragma unroll
    for (int e = 0; e < 8; ++e) o[e] = acc[e] * invl;
  }
}

// ---------------- context: mean(v) / cumsum(v) / scatter updates ------------
__global__ void vmean_k(const float* __restrict__ KVb, float* __restrict__ vm, int LK)
{
  int gid = blockIdx.x * blockDim.x + threadIdx.x;
  if (gid >= BATCH * NHEADS * 8) return;
  int d = gid & 63;         // h*8+e
  int b = gid >> 6;
  const float* vp = KVb + (size_t)b * LK * 128 + 64 + d;
  float s = 0.0f;
  for (int l = 0; l < LK; ++l) s += vp[(size_t)l * 128];
  vm[gid] = s / (float)LK;
}

__global__ void ctxmean_k(const float* __restrict__ vm, float* __restrict__ ctx, int LQ)
{
  int idx = blockIdx.x * blockDim.x + threadIdx.x;
  int total = BATCH * LQ * 64;
  if (idx >= total) return;
  int d = idx & 63;
  int b = (idx >> 6) / LQ;
  ctx[idx] = vm[b * 64 + d];
}

__global__ void ctxcum_k(const float* __restrict__ KVb, float* __restrict__ ctx, int L)
{
  int gid = blockIdx.x * blockDim.x + threadIdx.x;
  if (gid >= BATCH * NHEADS * 8) return;
  int d = gid & 63;
  int b = gid >> 6;
  const float* vp = KVb + (size_t)b * L * 128 + 64 + d;
  float* cp = ctx + (size_t)b * L * 64 + d;
  float run = 0.0f;
  for (int l = 0; l < L; ++l) { run += vp[(size_t)l * 128]; cp[(size_t)l * 64] = run; }
}

__global__ void scatter_k(const float* __restrict__ upd, const int* __restrict__ mtop,
                          float* __restrict__ ctx, int LQ, int ntop)
{
  int idx = blockIdx.x * blockDim.x + threadIdx.x;
  int total = BATCH * NHEADS * ntop * 8;
  if (idx >= total) return;
  int e  = idx & 7;
  int u  = (idx >> 3) % ntop;
  int bh = (idx >> 3) / ntop;
  int h = bh & 7;
  int b = bh >> 3;
  int qi = mtop[bh * ntop + u];
  ctx[((size_t)(b * LQ + qi)) * 64 + h * 8 + e] = upd[idx];
}

// ---------------- final projection of last token ----------------------------
__global__ void finalproj_k(const float* __restrict__ X, const float* __restrict__ pw,
                            const float* __restrict__ pb, float* __restrict__ out, int L)
{
  int idx = threadIdx.x;
  if (idx >= BATCH * 2) return;
  int b = idx >> 1, o = idx & 1;
  const float* x = X + ((size_t)b * L + (L - 1)) * 64;
  float acc = pb[o];
  for (int d = 0; d < 64; ++d) acc += x[d] * pw[o * 64 + d];
  out[idx] = acc;
}

// ===========================================================================
// Host orchestration
// ===========================================================================
struct AttnW { const float *wq,*bq,*wk,*bk,*wv,*bv,*wo,*bo; };
struct Scratch {
  float *QB, *KVB, *CTX, *MSC, *UPD, *VM, *WKV, *BKV;
  int *MTOP;
};

static inline unsigned gblk(long n) { return (unsigned)((n + 255) / 256); }

static void launch_gemm(hipStream_t st, const float* X, const float* W,
                        const float* bias, const float* resid, float* Y,
                        int T, int K, int N, int act)
{
  long waves = (long)(T / 16) * (N / 64);
  dim3 grid(gblk(waves * 32)), blk(256);
  if (K == 64 && N == 64)
    gemm_wmma_k<64, 64><<<grid, blk, 0, st>>>(X, W, bias, resid, Y, T, act);
  else if (K == 64 && N == 128)
    gemm_wmma_k<64, 128><<<grid, blk, 0, st>>>(X, W, bias, resid, Y, T, act);
  else if (K == 128 && N == 64)
    gemm_wmma_k<128, 64><<<grid, blk, 0, st>>>(X, W, bias, resid, Y, T, act);
  else if (K == 192 && N == 64)
    gemm_wmma_k<192, 64><<<grid, blk, 0, st>>>(X, W, bias, resid, Y, T, act);
}

static void launch_ln(hipStream_t st, const float* X, const float* g, const float* b,
                      float* Y, int T)
{
  layernorm_k<<<dim3(gblk((long)T * 32)), dim3(256), 0, st>>>(X, g, b, Y, T);
}

static void run_attention(hipStream_t st, const float* Xq, const float* Xkv,
                          int Lq, int Lk, const int* idx, int ns, int ntop, int mask,
                          const AttnW& w, float* Y, const Scratch& s)
{
  int Tq = BATCH * Lq, Tk = BATCH * Lk;
  pack_kv_k<<<dim3(32), dim3(256), 0, st>>>(w.wk, w.wv, w.bk, w.bv, s.WKV, s.BKV);
  launch_gemm(st, Xq,  w.wq,  w.bq,  nullptr, s.QB,  Tq, 64, 64,  0);
  launch_gemm(st, Xkv, s.WKV, s.BKV, nullptr, s.KVB, Tk, 64, 128, 0);
  mscore_k<<<dim3(gblk((long)BATCH * NHEADS * Lq)), dim3(256), 0, st>>>(
      s.QB, s.KVB, idx, s.MSC, Lq, Lk, ns);
  topk_k<<<dim3(BATCH * NHEADS), dim3(32), 0, st>>>(s.MSC, s.MTOP, Lq, ntop);
  probattn_k<<<dim3(gblk((long)BATCH * NHEADS * ntop * 32)), dim3(256), 0, st>>>(
      s.QB, s.KVB, s.MTOP, s.UPD, Lq, Lk, ntop, mask);
  if (mask) {
    ctxcum_k<<<dim3(gblk(BATCH * NHEADS * 8)), dim3(256), 0, st>>>(s.KVB, s.CTX, Lk);
  } else {
    vmean_k<<<dim3(gblk(BATCH * NHEADS * 8)), dim3(256), 0, st>>>(s.KVB, s.VM, Lk);
    ctxmean_k<<<dim3(gblk((long)BATCH * Lq * 64)), dim3(256), 0, st>>>(s.VM, s.CTX, Lq);
  }
  scatter_k<<<dim3(gblk((long)BATCH * NHEADS * ntop * 8)), dim3(256), 0, st>>>(
      s.UPD, s.MTOP, s.CTX, Lq, ntop);
  launch_gemm(st, s.CTX, w.wo, w.bo, Xq, Y, Tq, 64, 64, 0);   // + residual Xq
}

extern "C" void kernel_launch(void* const* d_in, const int* in_sizes, int n_in,
                              void* d_out, int out_size, void* d_ws, size_t ws_size,
                              hipStream_t stream)
{
  (void)in_sizes; (void)n_in; (void)out_size; (void)ws_size;
  auto F = [&](int i) { return (const float*)d_in[i]; };
  const float* x_enc = F(0);
  const float* x_dec = F(1);
  int p = 2;
  const float* enc_emb_w = F(p++);
  const float* dec_emb_w = F(p++);
  AttnW encA[2];
  const float *e_w1[2], *e_b1[2], *e_w2[2], *e_b2[2];
  const float *e_l1g[2], *e_l1b[2], *e_l2g[2], *e_l2b[2];
  for (int i = 0; i < 2; ++i) {
    encA[i].wq = F(p++); encA[i].bq = F(p++);
    encA[i].wk = F(p++); encA[i].bk = F(p++);
    encA[i].wv = F(p++); encA[i].bv = F(p++);
    encA[i].wo = F(p++); encA[i].bo = F(p++);
    e_w1[i] = F(p++); e_b1[i] = F(p++); e_w2[i] = F(p++); e_b2[i] = F(p++);
    e_l1g[i] = F(p++); e_l1b[i] = F(p++); e_l2g[i] = F(p++); e_l2b[i] = F(p++);
  }
  const float* cv_w = F(p++); const float* cv_b  = F(p++);
  const float* cv_g = F(p++); const float* cv_be = F(p++);
  const float* cv_rm = F(p++); const float* cv_rv = F(p++);
  const float* enc_ng = F(p++); const float* enc_nb = F(p++);
  AttnW dself, dcross;
  dself.wq = F(p++); dself.bq = F(p++); dself.wk = F(p++); dself.bk = F(p++);
  dself.wv = F(p++); dself.bv = F(p++); dself.wo = F(p++); dself.bo = F(p++);
  dcross.wq = F(p++); dcross.bq = F(p++); dcross.wk = F(p++); dcross.bk = F(p++);
  dcross.wv = F(p++); dcross.bv = F(p++); dcross.wo = F(p++); dcross.bo = F(p++);
  const float* d_w1 = F(p++); const float* d_b1 = F(p++);
  const float* d_w2 = F(p++); const float* d_b2 = F(p++);
  const float *d_l1g = F(p++), *d_l1b = F(p++);
  const float *d_l2g = F(p++), *d_l2b = F(p++);
  const float *d_l3g = F(p++), *d_l3b = F(p++);
  const float* dec_ng = F(p++); const float* dec_nb = F(p++);
  const float* proj_w = F(p++); const float* proj_b = F(p++);
  const int* idx_e1 = (const int*)d_in[p++];
  const int* idx_e2 = (const int*)d_in[p++];
  const int* idx_ds = (const int*)d_in[p++];
  const int* idx_dc = (const int*)d_in[p++];

  // ---------------- workspace arena (~342 MB) ----------------
  char* ws = (char*)d_ws;
  const size_t MB = 1ull << 20;
  float* X0  = (float*)(ws +   0 * MB);   // 32 MB  current activation
  float* QB  = (float*)(ws +  32 * MB);   // 32 MB  Q projection
  float* KVB = (float*)(ws +  64 * MB);   // 64 MB  fused K|V projection
  float* CTX = (float*)(ws + 128 * MB);   // 32 MB  attention context / conv out
  float* XA  = (float*)(ws + 160 * MB);   // 32 MB
  float* XN  = (float*)(ws + 192 * MB);   // 32 MB
  float* FF  = (float*)(ws + 224 * MB);   // 64 MB  FFN hidden
  float* X2  = (float*)(ws + 288 * MB);   // 32 MB
  float* ENC = (float*)(ws + 320 * MB);   // 16 MB  encoder output (persists)
  float* MSC = (float*)(ws + 336 * MB);   //  4 MB  sparsity scores
  int*   MT  = (int*)  (ws + 340 * MB);
  float* UPD = (float*)(ws + 340 * MB + (64 << 10));
  float* VM  = (float*)(ws + 340 * MB + (256 << 10));
  float* WKV = (float*)(ws + 341 * MB);
  float* BKV = (float*)(ws + 341 * MB + (64 << 10));
  float* WC  = (float*)(ws + 341 * MB + (128 << 10));
  float* BC  = (float*)(ws + 341 * MB + (192 << 10));
  float* IM2 = QB;                        // 96 MB im2col aliases dead QB+KVB

  Scratch sc{QB, KVB, CTX, MSC, UPD, VM, WKV, BKV, MT};

  // ===================== Encoder =====================
  embed_k<<<dim3(gblk((long)BATCH * LENC * 64)), dim3(256), 0, stream>>>(
      x_enc, enc_emb_w, X0, LENC);
  int T1 = BATCH * LENC;
  // layer 0 (L=4096, ns=ntop=ceil(ln 4096)=9)
  run_attention(stream, X0, X0, LENC, LENC, idx_e1, 9, 9, 0, encA[0], XA, sc);
  launch_ln(stream, XA, e_l1g[0], e_l1b[0], XN, T1);
  launch_gemm(stream, XN, e_w1[0], e_b1[0], nullptr, FF, T1, 64, 128, 1);  // GELU
  launch_gemm(stream, FF, e_w2[0], e_b2[0], XN, X2, T1, 128, 64, 0);
  launch_ln(stream, X2, e_l2g[0], e_l2b[0], X0, T1);
  // distilling conv (conv+BN folded into one WMMA GEMM, ELU epilogue) + pool
  pack_conv_k<<<dim3(gblk(64 * 192)), dim3(256), 0, stream>>>(
      cv_w, cv_b, cv_g, cv_be, cv_rm, cv_rv, WC, BC);
  im2col3_k<<<dim3(gblk((long)T1 * 192)), dim3(256), 0, stream>>>(X0, IM2, LENC);
  launch_gemm(stream, IM2, WC, BC, nullptr, CTX, T1, 192, 64, 2);          // ELU
  maxpool3s2_k<<<dim3(gblk((long)BATCH * (LENC / 2) * 64)), dim3(256), 0, stream>>>(
      CTX, X0, LENC);
  int L2 = LENC / 2;
  int T2 = BATCH * L2;
  // layer 1 (L=2048, ns=ntop=8)
  run_attention(stream, X0, X0, L2, L2, idx_e2, 8, 8, 0, encA[1], XA, sc);
  launch_ln(stream, XA, e_l1g[1], e_l1b[1], XN, T2);
  launch_gemm(stream, XN, e_w1[1], e_b1[1], nullptr, FF, T2, 64, 128, 1);
  launch_gemm(stream, FF, e_w2[1], e_b2[1], XN, X2, T2, 128, 64, 0);
  launch_ln(stream, X2, e_l2g[1], e_l2b[1], X0, T2);
  launch_ln(stream, X0, enc_ng, enc_nb, ENC, T2);

  // ===================== Decoder =====================
  int Td = BATCH * LDEC;
  embed_k<<<dim3(gblk((long)Td * 64)), dim3(256), 0, stream>>>(
      x_dec, dec_emb_w, X0, LDEC);
  // masked self-attention (cumsum context)
  run_attention(stream, X0, X0, LDEC, LDEC, idx_ds, 8, 8, 1, dself, XA, sc);
  launch_ln(stream, XA, d_l1g, d_l1b, XN, Td);
  // cross-attention vs encoder output (L_K = 2048)
  run_attention(stream, XN, ENC, LDEC, L2, idx_dc, 8, 8, 0, dcross, X2, sc);
  launch_ln(stream, X2, d_l2g, d_l2b, X0, Td);
  // FFN
  launch_gemm(stream, X0, d_w1, d_b1, nullptr, FF, Td, 64, 128, 1);
  launch_gemm(stream, FF, d_w2, d_b2, X0, XA, Td, 128, 64, 0);
  launch_ln(stream, XA, d_l3g, d_l3b, XN, Td);
  launch_ln(stream, XN, dec_ng, dec_nb, X2, Td);
  finalproj_k<<<dim3(1), dim3(64), 0, stream>>>(X2, proj_w, proj_b, (float*)d_out, LDEC);
}
